// FusionMoE_24395414241411
// MI455X (gfx1250) — matrix operand order repaired and checked by reading, not compiled, not run on hardware
//
#include <hip/hip_runtime.h>
#include <hip/hip_bf16.h>

typedef __attribute__((ext_vector_type(16))) _Float16 v16h;
typedef __attribute__((ext_vector_type(8)))  _Float16 v8h;
typedef __attribute__((ext_vector_type(8)))  float    v8f;

#define NB 16384
#define NE 4

__device__ __forceinline__ float gelu_exact(float v) {
    return 0.5f * v * (1.0f + erff(v * 0.70710678118654752f));
}

// ---- WMMA fragment helpers (CDNA5 16x16x32 f16, f32 accum) ----

// A: 16x32 f16 tile, row-major source (stride in halfs).
// lane L: row = r0 + (L&15); halves K {kk..kk+7},{kk+16..kk+23}, kk = k0 + 8*(L>=16)
__device__ __forceinline__ v16h load_a_frag(const _Float16* base, int stride, int r0, int k0) {
    int lane = threadIdx.x & 31;
    int row  = r0 + (lane & 15);
    int kk   = k0 + ((lane & 16) ? 8 : 0);
    const _Float16* p = base + (size_t)row * stride + kk;
    v8h lo = *(const v8h*)(p);
    v8h hi = *(const v8h*)(p + 16);
    v16h a;
#pragma unroll
    for (int i = 0; i < 8; ++i) { a[i] = lo[i]; a[i + 8] = hi[i]; }
    return a;
}

// Same A fragment but from an fp32 source, converting on the fly.
__device__ __forceinline__ v16h load_a_frag_f32(const float* base, int stride, int r0, int k0) {
    int lane = threadIdx.x & 31;
    int row  = r0 + (lane & 15);
    int kk   = k0 + ((lane & 16) ? 8 : 0);
    const float* p = base + (size_t)row * stride + kk;
    float4 q0 = *(const float4*)(p);
    float4 q1 = *(const float4*)(p + 4);
    float4 q2 = *(const float4*)(p + 16);
    float4 q3 = *(const float4*)(p + 20);
    v16h a;
    a[0]  = (_Float16)q0.x; a[1]  = (_Float16)q0.y; a[2]  = (_Float16)q0.z; a[3]  = (_Float16)q0.w;
    a[4]  = (_Float16)q1.x; a[5]  = (_Float16)q1.y; a[6]  = (_Float16)q1.z; a[7]  = (_Float16)q1.w;
    a[8]  = (_Float16)q2.x; a[9]  = (_Float16)q2.y; a[10] = (_Float16)q2.z; a[11] = (_Float16)q2.w;
    a[12] = (_Float16)q3.x; a[13] = (_Float16)q3.y; a[14] = (_Float16)q3.z; a[15] = (_Float16)q3.w;
    return a;
}

// B: 32x16 tile from transposed weights WT [N x K] row-major (strideK halfs).
__device__ __forceinline__ v16h load_b_frag(const _Float16* wt, int strideK, int n0, int k0) {
    int lane = threadIdx.x & 31;
    int col  = n0 + (lane & 15);
    int kk   = k0 + ((lane & 16) ? 8 : 0);
    const _Float16* p = wt + (size_t)col * strideK + kk;
    v8h lo = *(const v8h*)(p);
    v8h hi = *(const v8h*)(p + 16);
    v16h b;
#pragma unroll
    for (int i = 0; i < 8; ++i) { b[i] = lo[i]; b[i + 8] = hi[i]; }
    return b;
}

__device__ __forceinline__ v8f wmma16(v16h a, v16h b, v8f c) {
    return __builtin_amdgcn_wmma_f32_16x16x32_f16(false, a, false, b, (short)0, c, false, false);
}

// ---- Kernel 0: transpose fp32 [batch,K,N] -> f16 [batch,N,K] ----
__global__ __launch_bounds__(256) void k_transpose_h(const float* __restrict__ in,
                                                     _Float16* __restrict__ out,
                                                     int K, int N, long total) {
    long idx = (long)blockIdx.x * 256 + threadIdx.x;
    if (idx >= total) return;
    long kn = (long)K * N;
    long b  = idx / kn;
    long r  = idx - b * kn;
    int  k  = (int)(r / N);
    int  n  = (int)(r - (long)k * N);
    out[b * kn + (long)n * K + k] = (_Float16)in[idx];
}

// ---- Kernel 1: fusion GEMM + bias + LN + GELU + gate/top-2 routing ----
// block = 256 threads (8 waves), M=32 token rows, full 256 cols.
// wave w: col tiles {w*32, w*32+16} x row tiles {0,16} -> 4 accumulators.
__global__ __launch_bounds__(256) void k_fusion(const float* __restrict__ vis,
                                                const float* __restrict__ lang,
                                                const float* __restrict__ st,
                                                const float* __restrict__ bf,
                                                const float* __restrict__ gf,
                                                const float* __restrict__ bfln,
                                                const _Float16* __restrict__ WfT,
                                                const float* __restrict__ Wg,
                                                float* __restrict__ x32,
                                                _Float16* __restrict__ xh,
                                                float* __restrict__ wsel,
                                                float* __restrict__ psum,
                                                float* __restrict__ cnt) {
    __shared__ float sX[32 * 256];
    __shared__ float ps[256], ps2[256];
    __shared__ float smu[32], srs[32];
    __shared__ float sg[256 * 4];
    __shared__ float red[8];

    int t = threadIdx.x, lane = t & 31, w = t >> 5;
    int r0 = blockIdx.x * 32;
    int n0a = w * 32, n0b = w * 32 + 16;
    if (t < 8) red[t] = 0.f;

    v8f c[2][2] = {{{}, {}}, {{}, {}}};
    // straight-line K segments: vis [0,768), lang [768,1536), state [1536,1600)
#pragma unroll 2
    for (int ks = 0; ks < 24; ++ks) {
        int kl = ks * 32, kg = kl;
        v16h a0 = load_a_frag_f32(vis, 768, r0,      kl);
        v16h a1 = load_a_frag_f32(vis, 768, r0 + 16, kl);
        v16h b0 = load_b_frag(WfT, 1600, n0a, kg);
        v16h b1 = load_b_frag(WfT, 1600, n0b, kg);
        c[0][0] = wmma16(a0, b0, c[0][0]);
        c[0][1] = wmma16(a0, b1, c[0][1]);
        c[1][0] = wmma16(a1, b0, c[1][0]);
        c[1][1] = wmma16(a1, b1, c[1][1]);
    }
#pragma unroll 2
    for (int ks = 0; ks < 24; ++ks) {
        int kl = ks * 32, kg = 768 + kl;
        v16h a0 = load_a_frag_f32(lang, 768, r0,      kl);
        v16h a1 = load_a_frag_f32(lang, 768, r0 + 16, kl);
        v16h b0 = load_b_frag(WfT, 1600, n0a, kg);
        v16h b1 = load_b_frag(WfT, 1600, n0b, kg);
        c[0][0] = wmma16(a0, b0, c[0][0]);
        c[0][1] = wmma16(a0, b1, c[0][1]);
        c[1][0] = wmma16(a1, b0, c[1][0]);
        c[1][1] = wmma16(a1, b1, c[1][1]);
    }
#pragma unroll
    for (int ks = 0; ks < 2; ++ks) {
        int kl = ks * 32, kg = 1536 + kl;
        v16h a0 = load_a_frag_f32(st, 64, r0,      kl);
        v16h a1 = load_a_frag_f32(st, 64, r0 + 16, kl);
        v16h b0 = load_b_frag(WfT, 1600, n0a, kg);
        v16h b1 = load_b_frag(WfT, 1600, n0b, kg);
        c[0][0] = wmma16(a0, b0, c[0][0]);
        c[0][1] = wmma16(a0, b1, c[0][1]);
        c[1][0] = wmma16(a1, b0, c[1][0]);
        c[1][1] = wmma16(a1, b1, c[1][1]);
    }
    {   // spill accum + bias to LDS for LayerNorm
        int col = lane & 15, rb = (lane & 16) ? 8 : 0;
#pragma unroll
        for (int m = 0; m < 2; ++m)
#pragma unroll
            for (int v = 0; v < 8; ++v) {
                int row = m * 16 + v + rb;
                sX[row * 256 + n0a + col] = c[m][0][v] + bf[n0a + col];
                sX[row * 256 + n0b + col] = c[m][1][v] + bf[n0b + col];
            }
    }
    __syncthreads();
    int row = t >> 3, seg = t & 7;          // 8 threads per row, 32 elems each
    float s1 = 0.f, s2 = 0.f;
#pragma unroll
    for (int i = 0; i < 32; ++i) {
        float v = sX[row * 256 + seg * 32 + i];
        s1 += v; s2 += v * v;
    }
    ps[t] = s1; ps2[t] = s2;
    __syncthreads();
    if (t < 32) {
        float a1 = 0.f, a2 = 0.f;
#pragma unroll
        for (int i = 0; i < 8; ++i) { a1 += ps[t * 8 + i]; a2 += ps2[t * 8 + i]; }
        float mu  = a1 * (1.0f / 256.0f);
        float var = a2 * (1.0f / 256.0f) - mu * mu;
        smu[t] = mu; srs[t] = rsqrtf(var + 1e-5f);
    }
    __syncthreads();
    float mu = smu[row], rs = srs[row];
    float lg[4] = {0.f, 0.f, 0.f, 0.f};
#pragma unroll
    for (int i = 0; i < 32; ++i) {
        int cg = seg * 32 + i;
        float v = sX[row * 256 + cg];
        float g = gelu_exact(gf[cg] * (v - mu) * rs + bfln[cg]);
        x32[(size_t)(r0 + row) * 256 + cg] = g;
        xh [(size_t)(r0 + row) * 256 + cg] = (_Float16)g;
        float4 wg = *(const float4*)(Wg + cg * 4);
        lg[0] += g * wg.x; lg[1] += g * wg.y; lg[2] += g * wg.z; lg[3] += g * wg.w;
    }
#pragma unroll
    for (int e = 0; e < 4; ++e) sg[t * 4 + e] = lg[e];
    __syncthreads();
    if (t < 32) {   // one thread per row: softmax + top-2 + stats
        float l[4] = {0.f, 0.f, 0.f, 0.f};
#pragma unroll
        for (int s = 0; s < 8; ++s)
#pragma unroll
            for (int e = 0; e < 4; ++e) l[e] += sg[(t * 8 + s) * 4 + e];
        float m = fmaxf(fmaxf(l[0], l[1]), fmaxf(l[2], l[3]));
        float p[4], ssum = 0.f;
#pragma unroll
        for (int e = 0; e < 4; ++e) { p[e] = expf(l[e] - m); ssum += p[e]; }
        float inv = 1.0f / ssum;
        int i0 = 0;
#pragma unroll
        for (int e = 1; e < 4; ++e) if (l[e] > l[i0]) i0 = e;
        int i1 = (i0 == 0) ? 1 : 0;
#pragma unroll
        for (int e = 0; e < 4; ++e) if (e != i0 && l[e] > l[i1]) i1 = e;
        float tt = expf(l[i1] - l[i0]);
        float w0 = 1.0f / (1.0f + tt), w1 = tt * w0;
#pragma unroll
        for (int e = 0; e < 4; ++e)
            wsel[(size_t)(r0 + t) * 4 + e] = (e == i0) ? w0 : ((e == i1) ? w1 : 0.f);
#pragma unroll
        for (int e = 0; e < 4; ++e) atomicAdd(&red[e], p[e] * inv);
        atomicAdd(&red[4 + i0], 1.0f);
        atomicAdd(&red[4 + i1], 1.0f);
    }
    __syncthreads();
    if (t < 4) { atomicAdd(&psum[t], red[t]); atomicAdd(&cnt[t], red[4 + t]); }
}

__global__ void k_zero8(float* p) { if (threadIdx.x < 8) p[threadIdx.x] = 0.f; }

__global__ void k_lb(const float* psum, const float* cnt, float* lb) {
    if (threadIdx.x == 0) {
        float acc = 0.f;
        for (int e = 0; e < 4; ++e)
            acc += (cnt[e] / ((float)NB * 2.0f)) * (psum[e] / (float)NB);
        lb[0] = 4.0f * acc;
    }
}

// ---- Kernel 2: experts (dense) + residual LN + weighted combine ----
// block = 256 threads, M=32 token rows, iterate all 4 experts.
// sr (f32 pre-LN) overlays sh1 (dead after stage 2) to stay under 64KB LDS.
__global__ __launch_bounds__(256) void k_experts(const _Float16* __restrict__ xh,
                                                 const float* __restrict__ x32,
                                                 const _Float16* __restrict__ W1T,
                                                 const float* __restrict__ b1,
                                                 const _Float16* __restrict__ W2T,
                                                 const float* __restrict__ b2,
                                                 const _Float16* __restrict__ W3T,
                                                 const float* __restrict__ b3,
                                                 const float* __restrict__ eg,
                                                 const float* __restrict__ eb,
                                                 const float* __restrict__ wsel,
                                                 float* __restrict__ out) {
    __shared__ __align__(16) char sbuf[32 * 512 * 2];    // 32KB overlay
    _Float16* sh1 = (_Float16*)sbuf;                     // [32][512] f16
    float*    sr  = (float*)sbuf;                        // [32][256] f32
    __shared__ _Float16 sh2[32 * 256];                   // 16KB
    __shared__ float ps[256], ps2[256];
    __shared__ float smu[32], srs[32];

    int t = threadIdx.x, lane = t & 31, w = t >> 5;
    int r0  = blockIdx.x * 32;
    int row = t >> 3, seg = t & 7;
    int col = lane & 15, rb = (lane & 16) ? 8 : 0;

    float acc[32];
#pragma unroll
    for (int i = 0; i < 32; ++i) acc[i] = 0.f;

    for (int e = 0; e < 4; ++e) {
        const _Float16* w1t = W1T + (size_t)e * 512 * 256;
        const _Float16* w2t = W2T + (size_t)e * 256 * 512;
        const _Float16* w3t = W3T + (size_t)e * 256 * 256;
        const float* b1e = b1 + e * 512;
        const float* b2e = b2 + e * 256;
        const float* b3e = b3 + e * 256;
        __builtin_prefetch(w1t + (size_t)lane * 256, 0, 1);

        // stage 1: h1 = gelu(x @ W1 + b1)  [32,512], K=256 ; 4 col x 2 row tiles/wave
        {
            v8f c[2][4] = {{{}, {}, {}, {}}, {{}, {}, {}, {}}};
            for (int kt = 0; kt < 8; ++kt) {
                int k0 = kt * 32;
                v16h a0 = load_a_frag(xh, 256, r0,      k0);
                v16h a1 = load_a_frag(xh, 256, r0 + 16, k0);
#pragma unroll
                for (int j = 0; j < 4; ++j) {
                    v16h b = load_b_frag(w1t, 256, (w * 4 + j) * 16, k0);
                    c[0][j] = wmma16(a0, b, c[0][j]);
                    c[1][j] = wmma16(a1, b, c[1][j]);
                }
            }
#pragma unroll
            for (int m = 0; m < 2; ++m)
#pragma unroll
                for (int j = 0; j < 4; ++j) {
                    int cg = (w * 4 + j) * 16 + col;
#pragma unroll
                    for (int v = 0; v < 8; ++v)
                        sh1[(m * 16 + v + rb) * 512 + cg] =
                            (_Float16)gelu_exact(c[m][j][v] + b1e[cg]);
                }
        }
        __syncthreads();

        // stage 2: h2 = gelu(h1 @ W2 + b2)  [32,256], K=512 ; 2 col x 2 row tiles/wave
        {
            v8f d[2][2] = {{{}, {}}, {{}, {}}};
            for (int kt = 0; kt < 16; ++kt) {
                int k0 = kt * 32;
                v16h a0 = load_a_frag(sh1, 512, 0,  k0);
                v16h a1 = load_a_frag(sh1, 512, 16, k0);
#pragma unroll
                for (int j = 0; j < 2; ++j) {
                    v16h b = load_b_frag(w2t, 512, (w * 2 + j) * 16, k0);
                    d[0][j] = wmma16(a0, b, d[0][j]);
                    d[1][j] = wmma16(a1, b, d[1][j]);
                }
            }
            __syncthreads();    // all sh1 reads done before sr overlay is written
#pragma unroll
            for (int m = 0; m < 2; ++m)
#pragma unroll
                for (int j = 0; j < 2; ++j) {
                    int cg = (w * 2 + j) * 16 + col;
#pragma unroll
                    for (int v = 0; v < 8; ++v)
                        sh2[(m * 16 + v + rb) * 256 + cg] =
                            (_Float16)gelu_exact(d[m][j][v] + b2e[cg]);
                }
        }
        __syncthreads();

        // stage 3: r = x + h2 @ W3 + b3   [32,256], K=256 ; 2 col x 2 row tiles/wave
        {
            v8f f[2][2] = {{{}, {}}, {{}, {}}};
            for (int kt = 0; kt < 8; ++kt) {
                int k0 = kt * 32;
                v16h a0 = load_a_frag(sh2, 256, 0,  k0);
                v16h a1 = load_a_frag(sh2, 256, 16, k0);
#pragma unroll
                for (int j = 0; j < 2; ++j) {
                    v16h b = load_b_frag(w3t, 256, (w * 2 + j) * 16, k0);
                    f[0][j] = wmma16(a0, b, f[0][j]);
                    f[1][j] = wmma16(a1, b, f[1][j]);
                }
            }
#pragma unroll
            for (int m = 0; m < 2; ++m)
#pragma unroll
                for (int j = 0; j < 2; ++j) {
                    int cg = (w * 2 + j) * 16 + col;
#pragma unroll
                    for (int v = 0; v < 8; ++v) {
                        int rr = m * 16 + v + rb;
                        sr[rr * 256 + cg] =
                            f[m][j][v] + b3e[cg] + x32[(size_t)(r0 + rr) * 256 + cg];
                    }
                }
        }
        __syncthreads();

        // LayerNorm(sr) with eg/eb, then acc += wsel * y
        float s1 = 0.f, s2 = 0.f;
#pragma unroll
        for (int i = 0; i < 32; ++i) {
            float v = sr[row * 256 + seg * 32 + i];
            s1 += v; s2 += v * v;
        }
        ps[t] = s1; ps2[t] = s2;
        __syncthreads();
        if (t < 32) {
            float a1 = 0.f, a2 = 0.f;
#pragma unroll
            for (int i = 0; i < 8; ++i) { a1 += ps[t * 8 + i]; a2 += ps2[t * 8 + i]; }
            float mu  = a1 * (1.0f / 256.0f);
            float var = a2 * (1.0f / 256.0f) - mu * mu;
            smu[t] = mu; srs[t] = rsqrtf(var + 1e-5f);
        }
        __syncthreads();
        float mu = smu[row], rs = srs[row];
        float wse = wsel[(size_t)(r0 + row) * 4 + e];
        const float* ege = eg + e * 256;
        const float* ebe = eb + e * 256;
#pragma unroll
        for (int i = 0; i < 32; ++i) {
            int cg = seg * 32 + i;
            float v = sr[row * 256 + cg];
            acc[i] += wse * (ege[cg] * (v - mu) * rs + ebe[cg]);
        }
        __syncthreads();   // sr/sh1 overlay reused next expert
    }
#pragma unroll
    for (int i = 0; i < 32; ++i)
        out[(size_t)(r0 + row) * 256 + seg * 32 + i] = acc[i];
}

extern "C" void kernel_launch(void* const* d_in, const int* in_sizes, int n_in,
                              void* d_out, int out_size, void* d_ws, size_t ws_size,
                              hipStream_t stream) {
    const float* vis  = (const float*)d_in[0];
    const float* lang = (const float*)d_in[1];
    const float* st   = (const float*)d_in[2];
    const float* Wf   = (const float*)d_in[3];
    const float* bf   = (const float*)d_in[4];
    const float* gf   = (const float*)d_in[5];
    const float* bfln = (const float*)d_in[6];
    const float* Wg   = (const float*)d_in[7];
    const float* W1   = (const float*)d_in[8];
    const float* b1   = (const float*)d_in[9];
    const float* W2   = (const float*)d_in[10];
    const float* b2   = (const float*)d_in[11];
    const float* W3   = (const float*)d_in[12];
    const float* b3   = (const float*)d_in[13];
    const float* eg   = (const float*)d_in[14];
    const float* eb   = (const float*)d_in[15];
    float* out = (float*)d_out;

    char* ws = (char*)d_ws;
    size_t off = 0;
    auto alloc = [&](size_t bytes) -> void* {
        void* p = ws + off;
        off += (bytes + 255) & ~(size_t)255;
        return p;
    };
    _Float16* WfT = (_Float16*)alloc((size_t)1600 * 256 * 2);
    _Float16* W1T = (_Float16*)alloc((size_t)NE * 256 * 512 * 2);
    _Float16* W2T = (_Float16*)alloc((size_t)NE * 512 * 256 * 2);
    _Float16* W3T = (_Float16*)alloc((size_t)NE * 256 * 256 * 2);
    _Float16* xh  = (_Float16*)alloc((size_t)NB * 256 * 2);
    float*    x32 = (float*)alloc((size_t)NB * 256 * 4);
    float*    wsel= (float*)alloc((size_t)NB * 4 * 4);
    float*    stats = (float*)alloc(8 * 4);   // psum[4], cnt[4]

    long tWf = 1600L * 256, tW1 = 4L * 256 * 512, tW2 = 4L * 512 * 256, tW3 = 4L * 256 * 256;
    k_transpose_h<<<(int)((tWf + 255) / 256), 256, 0, stream>>>(Wf, WfT, 1600, 256, tWf);
    k_transpose_h<<<(int)((tW1 + 255) / 256), 256, 0, stream>>>(W1, W1T, 256, 512, tW1);
    k_transpose_h<<<(int)((tW2 + 255) / 256), 256, 0, stream>>>(W2, W2T, 512, 256, tW2);
    k_transpose_h<<<(int)((tW3 + 255) / 256), 256, 0, stream>>>(W3, W3T, 256, 256, tW3);

    k_zero8<<<1, 8, 0, stream>>>(stats);

    k_fusion<<<NB / 32, 256, 0, stream>>>(vis, lang, st, bf, gf, bfln, WfT, Wg,
                                          x32, xh, wsel, stats, stats + 4);

    k_experts<<<NB / 32, 256, 0, stream>>>(xh, x32, W1T, b1, W2T, b2, W3T, b3,
                                           eg, eb, wsel, out);

    k_lb<<<1, 32, 0, stream>>>(stats, stats + 4, out + (size_t)NB * 256);
}
